// Selection_65962107732500
// MI455X (gfx1250) — matrix operand order repaired and checked by reading, not compiled, not run on hardware
//
#include <hip/hip_runtime.h>
#include <hip/hip_bf16.h>

#define N_EXPERTS 8
#define D_IN 2048
#define D_OUT 2048
#define BATCH 8192

#define TM 128
#define TN 128
#define TK 32

typedef __attribute__((ext_vector_type(16))) __bf16 v16bf;
typedef __attribute__((ext_vector_type(8)))  float  v8f;

union FragBF { uint4 u[2]; v16bf v; };

// round-to-nearest-even f32 -> bf16 (bit trick; NaN not expected in this workload)
__device__ __forceinline__ unsigned short bf16_rne(float f) {
    unsigned u = __float_as_uint(f);
    u += 0x7FFFu + ((u >> 16) & 1u);
    return (unsigned short)(u >> 16);
}
__device__ __forceinline__ float bf16_to_f32(unsigned short h) {
    return __uint_as_float(((unsigned)h) << 16);
}

// ---------------- kernel 0: zero per-expert counters ----------------
__global__ void moe_init(int* counts) {
    if (threadIdx.x < N_EXPERTS) counts[threadIdx.x] = 0;
}

// ---------------- kernel 1: scatter tokens into per-expert buckets --
// Also emits the tuple tail: (float)mxs, (float)actions.
__global__ void moe_scatter(const int* __restrict__ actions,
                            const int* __restrict__ mxs,
                            int* __restrict__ counts,
                            int* __restrict__ bucket,
                            float* __restrict__ out_tail) {
    int i = blockIdx.x * blockDim.x + threadIdx.x;
    if (i < BATCH) {
        int e = actions[i];
        int pos = atomicAdd(&counts[e], 1);
        bucket[e * BATCH + pos] = i;
        out_tail[i]         = (float)mxs[i];
        out_tail[BATCH + i] = (float)actions[i];
    }
}

// ---------------- kernel 2: grouped GEMM per expert -----------------
// Tile: 128(M tokens) x 128(N features), K-step 32. 8 waves, each owns
// a 64x32 sub-tile = 4x2 WMMA accumulators. bf16x3 fp32 emulation.
__global__ __launch_bounds__(256) void moe_gemm(
        const float* __restrict__ xs,     // [BATCH, D_IN]
        const float* __restrict__ W,      // [E, D_IN, D_OUT]
        const float* __restrict__ bias,   // [E, D_OUT]
        const int*   __restrict__ counts, // [E]
        const int*   __restrict__ bucket, // [E, BATCH]
        float*       __restrict__ out)    // [BATCH, D_OUT]
{
    const int e      = blockIdx.z;
    const int cnt    = counts[e];
    const int m_base = blockIdx.y * TM;
    if (m_base >= cnt) return;              // uniform early-exit (EXEC stays full)
    const int n_base = blockIdx.x * TN;

    __shared__ __align__(16) unsigned short Ahi[TM * TK];
    __shared__ __align__(16) unsigned short Alo[TM * TK];
    __shared__ __align__(16) unsigned short Bhi[TN * TK];
    __shared__ __align__(16) unsigned short Blo[TN * TK];

    const int tid  = threadIdx.x;
    const int lane = tid & 31;
    const int wid  = tid >> 5;

    // ---- staging assignments ----
    // A: 2 threads per row, 16 k-elements each (4 x float4)
    const int ar = tid >> 1;            // 0..127 (tile row)
    const int ac = (tid & 1) * 16;      // k sub-offset 0 or 16
    const bool rowok = (m_base + ar) < cnt;
    const int tokenA = rowok ? bucket[e * BATCH + m_base + ar] : 0;
    const float* xrow = xs + (size_t)tokenA * D_IN;

    // B: 2 threads per column, 16 k-elements each (coalesced across n)
    const int bn = tid & 127;           // tile column (feature)
    const int bk = (tid >> 7) * 16;     // k sub-offset 0 or 16
    const float* wcol = W + (size_t)e * D_IN * D_OUT + (n_base + bn);

    // ---- wave sub-tile ----
    const int wm = (wid & 1) * 64;      // wave M origin within tile
    const int wn = (wid >> 1) * 32;     // wave N origin within tile

    v8f acc[4][2] = {};

    for (int kb = 0; kb < D_IN; kb += TK) {
        // ----- stage A tile (fp32 -> bf16 hi/lo), row-major [m][k] -----
        #pragma unroll
        for (int j = 0; j < 4; ++j) {
            float4 f;
            if (rowok) f = *(const float4*)(xrow + kb + ac + 4 * j);
            else       f = make_float4(0.f, 0.f, 0.f, 0.f);
            unsigned short h0 = bf16_rne(f.x), h1 = bf16_rne(f.y);
            unsigned short h2 = bf16_rne(f.z), h3 = bf16_rne(f.w);
            unsigned short l0 = bf16_rne(f.x - bf16_to_f32(h0));
            unsigned short l1 = bf16_rne(f.y - bf16_to_f32(h1));
            unsigned short l2 = bf16_rne(f.z - bf16_to_f32(h2));
            unsigned short l3 = bf16_rne(f.w - bf16_to_f32(h3));
            int idx = ar * TK + ac + 4 * j;
            *(uint2*)&Ahi[idx] = make_uint2((unsigned)h0 | ((unsigned)h1 << 16),
                                            (unsigned)h2 | ((unsigned)h3 << 16));
            *(uint2*)&Alo[idx] = make_uint2((unsigned)l0 | ((unsigned)l1 << 16),
                                            (unsigned)l2 | ((unsigned)l3 << 16));
        }

        // ----- stage B tile transposed: Bt[n][k] (fp32 -> bf16 hi/lo) -----
        {
            unsigned hp[8], lp[8];
            #pragma unroll
            for (int j = 0; j < 16; ++j) {
                float w = wcol[(size_t)(kb + bk + j) * D_OUT];
                unsigned short h = bf16_rne(w);
                unsigned short l = bf16_rne(w - bf16_to_f32(h));
                if (j & 1) { hp[j >> 1] |= (unsigned)h << 16; lp[j >> 1] |= (unsigned)l << 16; }
                else       { hp[j >> 1]  = (unsigned)h;       lp[j >> 1]  = (unsigned)l;       }
            }
            int idx = bn * TK + bk;
            *(uint4*)&Bhi[idx]     = make_uint4(hp[0], hp[1], hp[2], hp[3]);
            *(uint4*)&Bhi[idx + 8] = make_uint4(hp[4], hp[5], hp[6], hp[7]);
            *(uint4*)&Blo[idx]     = make_uint4(lp[0], lp[1], lp[2], lp[3]);
            *(uint4*)&Blo[idx + 8] = make_uint4(lp[4], lp[5], lp[6], lp[7]);
        }

        __syncthreads();

        // ----- B fragments (lane: N = L&15, K = h*8..+7 and h*8+16..+23) -----
        FragBF bh[2], bl[2];
        #pragma unroll
        for (int nf = 0; nf < 2; ++nf) {
            int n  = wn + nf * 16 + (lane & 15);
            int kh = (lane >> 4) * 8;
            const uint4* ph = (const uint4*)&Bhi[n * TK + kh];
            const uint4* pl = (const uint4*)&Blo[n * TK + kh];
            bh[nf].u[0] = ph[0]; bh[nf].u[1] = *(const uint4*)&Bhi[n * TK + kh + 16];
            bl[nf].u[0] = pl[0]; bl[nf].u[1] = *(const uint4*)&Blo[n * TK + kh + 16];
        }

        // ----- A fragments + WMMA (lane: M = L&15, K = h*8..+7 and h*8+16..+23) -----
        #pragma unroll
        for (int mf = 0; mf < 4; ++mf) {
            int m  = wm + mf * 16 + (lane & 15);
            int kh = (lane >> 4) * 8;
            FragBF ah, al;
            ah.u[0] = *(const uint4*)&Ahi[m * TK + kh];
            ah.u[1] = *(const uint4*)&Ahi[m * TK + kh + 16];
            al.u[0] = *(const uint4*)&Alo[m * TK + kh];
            al.u[1] = *(const uint4*)&Alo[m * TK + kh + 16];
            #pragma unroll
            for (int nf = 0; nf < 2; ++nf) {
                acc[mf][nf] = __builtin_amdgcn_wmma_f32_16x16x32_bf16(
                    false, ah.v, false, bh[nf].v, (short)0, acc[mf][nf], false, false);
                acc[mf][nf] = __builtin_amdgcn_wmma_f32_16x16x32_bf16(
                    false, ah.v, false, bl[nf].v, (short)0, acc[mf][nf], false, false);
                acc[mf][nf] = __builtin_amdgcn_wmma_f32_16x16x32_bf16(
                    false, al.v, false, bh[nf].v, (short)0, acc[mf][nf], false, false);
            }
        }

        __syncthreads();
    }

    // ----- epilogue: add routed bias, scatter rows through bucket -----
    #pragma unroll
    for (int nf = 0; nf < 2; ++nf) {
        int n = n_base + wn + nf * 16 + (lane & 15);
        float bv = bias[e * D_OUT + n];
        #pragma unroll
        for (int mf = 0; mf < 4; ++mf) {
            #pragma unroll
            for (int r = 0; r < 8; ++r) {
                int ml = wm + mf * 16 + r + (lane >> 4) * 8; // C layout: VGPR r -> M=r (+8 hi half)
                int gm = m_base + ml;
                if (gm < cnt) {
                    int token = bucket[e * BATCH + gm];
                    out[(size_t)token * D_OUT + n] = acc[mf][nf][r] + bv;
                }
            }
        }
    }
}

extern "C" void kernel_launch(void* const* d_in, const int* in_sizes, int n_in,
                              void* d_out, int out_size, void* d_ws, size_t ws_size,
                              hipStream_t stream) {
    (void)in_sizes; (void)n_in; (void)out_size; (void)ws_size;
    const float* xs      = (const float*)d_in[0];
    const float* W       = (const float*)d_in[1];
    const float* bias    = (const float*)d_in[2];
    const int*   mxs     = (const int*)d_in[3];
    const int*   actions = (const int*)d_in[4];
    float* out = (float*)d_out;

    int* counts = (int*)d_ws;          // 8 ints (padded to 256 B)
    int* bucket = counts + 64;         // [E, BATCH] token indices = 256 KB

    moe_init<<<1, 32, 0, stream>>>(counts);
    moe_scatter<<<BATCH / 256, 256, 0, stream>>>(actions, mxs, counts, bucket,
                                                 out + (size_t)BATCH * D_OUT);

    dim3 grid(D_OUT / TN, BATCH / TM, N_EXPERTS);   // worst-case m-tiles; early-exit per expert
    moe_gemm<<<grid, 256, 0, stream>>>(xs, W, bias, counts, bucket, out);
}
